// Transformer_57071525429444
// MI455X (gfx1250) — matrix operand order repaired and checked by reading, not compiled, not run on hardware
//
#include <hip/hip_runtime.h>
#include <hip/hip_bf16.h>
#include <stdint.h>

typedef __bf16 bf16_t;
typedef __attribute__((ext_vector_type(16))) __bf16 v16bf;
typedef __attribute__((ext_vector_type(8)))  __bf16 v8bf;
typedef __attribute__((ext_vector_type(8)))  float  v8f;

union BfFrag { v16bf v; v8bf h[2]; };

enum GemmFlags {
  F_OUTF32  = 1,   // store f32 result to Cf
  F_ADDRES  = 2,   // Cf += result (residual), implies f32 store
  F_BIAS    = 4,   // add bias[n]
  F_RELU    = 8,   // relu before store
  F_OUTB16  = 16,  // store bf16 row-major to Cb
  F_OUTB16T = 32   // store bf16 transposed: Cb[n*ldcb + m]
};

// ---------------------------------------------------------------------------
// Generic GEMM: D[M,N] = A[M,K](bf16,row) * Bt[N,K](bf16,row == B^T) (+epilogue)
// Block: 256 threads (8 waves). Tile: 128x128, K-step 32.
// Each wave: 16(M) x 128(N) via 8 accumulators of v_wmma_f32_16x16x32_bf16.
// B tile is brought into LDS with GLOBAL_LOAD_ASYNC_TO_LDS_B128 (ASYNCcnt).
// The LDS offset operand is derived from ldsB's own flat address (low 32 bits
// of a generic pointer into the LDS aperture == wave-relative LDS offset per
// ISA 10.2), which also escapes ldsB so the asm's memory clobber is honored.
// M, N multiples of 128; K multiple of 32 (guaranteed by caller).
// ---------------------------------------------------------------------------
__global__ void __launch_bounds__(256)
gemm_bf16_wmma(const bf16_t* __restrict__ A, int lda,
               const bf16_t* __restrict__ Bt, int ldb,
               float* __restrict__ Cf, long long ldc,
               bf16_t* __restrict__ Cb, long long ldcb,
               const float* __restrict__ bias,
               int K, int flags)
{
  __shared__ __attribute__((aligned(16))) bf16_t ldsB[128 * 32];

  const int tid  = threadIdx.x;
  const int lane = tid & 31;
  const int wv   = tid >> 5;
  const int n0   = blockIdx.x * 128;
  const int m0   = blockIdx.y * 128 + wv * 16;
  const int mr   = lane & 15;   // row (A) / col (B,C) within 16-tile
  const int hf   = lane >> 4;   // lane half selects K/M half per ISA layout

  v8f acc[8] = {};

  // A fragment: lane(0-15) holds M=mr, K = hf*8 + [0..7] and 16+hf*8 + [0..7]
  const bf16_t* aRow  = A  + (long long)(m0 + mr) * lda + hf * 8;
  // cooperative async LDS fill: thread t supplies 32 consecutive bytes of
  // B-tile row (t>>1).
  const bf16_t* bFill = Bt + (long long)(n0 + (tid >> 1)) * ldb + (tid & 1) * 16;
  // LDS byte offset for this thread's fill target; ptrtoint escapes ldsB.
  const unsigned ldsOff = (unsigned)(unsigned long long)(const void*)ldsB
                        + (unsigned)((tid >> 1) * 64 + (tid & 1) * 32);

  for (int k0 = 0; k0 < K; k0 += 32) {
    __syncthreads();  // previous iteration's LDS reads complete
    {
      const bf16_t* src = bFill + k0;
      asm volatile(
          "global_load_async_to_lds_b128 %0, %1, off\n\t"
          "global_load_async_to_lds_b128 %0, %1, off offset:16\n\t"
          "s_wait_asynccnt 0x0"
          :
          : "v"(ldsOff), "v"(src)
          : "memory");
    }
    __syncthreads();  // all waves' async fills visible

    BfFrag a;
    a.h[0] = *(const v8bf*)(aRow + k0);
    a.h[1] = *(const v8bf*)(aRow + k0 + 16);
    __builtin_prefetch(aRow + k0 + 32, 0, 1);  // global_prefetch_b8

    // Preload all 8 B fragments (distinct regs) -> one ds_load clause,
    // then 8 back-to-back WMMAs.
    BfFrag bb[8];
#pragma unroll
    for (int j = 0; j < 8; ++j) {
      // B fragment: lane holds N=mr, K = hf*8 + [0..7] and 16+hf*8 + [0..7]
      const bf16_t* bp = ldsB + (j * 16 + mr) * 32 + hf * 8;
      bb[j].h[0] = *(const v8bf*)bp;
      bb[j].h[1] = *(const v8bf*)(bp + 16);
    }
#pragma unroll
    for (int j = 0; j < 8; ++j) {
      acc[j] = __builtin_amdgcn_wmma_f32_16x16x32_bf16(
          false, a.v, false, bb[j].v, (short)0, acc[j], false, false);
    }
  }

  // Writeback: VGPR r -> M = r + hf*8; N = lane&15 within 16-tile
#pragma unroll
  for (int j = 0; j < 8; ++j) {
    const int n = n0 + j * 16 + mr;
    const float bv = (flags & F_BIAS) ? bias[n] : 0.0f;
#pragma unroll
    for (int r = 0; r < 8; ++r) {
      const int m = m0 + r + hf * 8;
      float v = acc[j][r] + bv;
      if (flags & F_RELU)   v = fmaxf(v, 0.0f);
      if (flags & F_ADDRES) v += Cf[(long long)m * ldc + n];
      if (flags & (F_OUTF32 | F_ADDRES)) Cf[(long long)m * ldc + n] = v;
      if (flags & F_OUTB16)  Cb[(long long)m * ldcb + n] = (bf16_t)v;
      if (flags & F_OUTB16T) Cb[(long long)n * ldcb + m] = (bf16_t)v;
    }
  }
}

// ---------------------------------------------------------------------------
// Batched transpose + f32->bf16: dst[b][c][r] = src[b][r][c]
// ---------------------------------------------------------------------------
__global__ void __launch_bounds__(256)
tconv(const float* __restrict__ src, bf16_t* __restrict__ dst,
      int batch, int R, int Cc)
{
  long long total = (long long)batch * R * Cc;
  long long i = (long long)blockIdx.x * 256 + threadIdx.x;
  if (i >= total) return;
  long long per = (long long)R * Cc;
  long long b = i / per, rem = i % per;
  long long r = rem / Cc, c = rem % Cc;
  dst[b * per + c * R + r] = (bf16_t)src[i];
}

// ---------------------------------------------------------------------------
// x[b,t,:] = embed[tok[b,t],:] + pos[t,:]
// ---------------------------------------------------------------------------
__global__ void __launch_bounds__(256)
embed_add(const int* __restrict__ tok, const float* __restrict__ emb,
          const float* __restrict__ pos, float* __restrict__ X,
          int T_, int C_, int total)
{
  int i = blockIdx.x * 256 + threadIdx.x;
  if (i >= total) return;
  int c  = i % C_;
  int bt = i / C_;
  int t  = bt % T_;
  X[i] = emb[(long long)tok[bt] * C_ + c] + pos[(long long)t * C_ + c];
}

// ---------------------------------------------------------------------------
// In-place LayerNorm over C=256; one wave32 per row; also emit bf16 copy.
// ---------------------------------------------------------------------------
__global__ void __launch_bounds__(256)
layernorm_rows(float* __restrict__ X, bf16_t* __restrict__ Xb,
               const float* __restrict__ g, const float* __restrict__ bta)
{
  const int lane = threadIdx.x & 31;
  const int row  = blockIdx.x * 8 + (threadIdx.x >> 5);
  float* xr = X + (long long)row * 256;

  float v[8];
#pragma unroll
  for (int i = 0; i < 8; ++i) v[i] = xr[lane * 8 + i];

  float s = 0.f;
#pragma unroll
  for (int i = 0; i < 8; ++i) s += v[i];
#pragma unroll
  for (int o = 16; o > 0; o >>= 1) s += __shfl_xor(s, o, 32);
  const float mean = s * (1.0f / 256.0f);

  float q = 0.f;
#pragma unroll
  for (int i = 0; i < 8; ++i) { float d = v[i] - mean; q += d * d; }
#pragma unroll
  for (int o = 16; o > 0; o >>= 1) q += __shfl_xor(q, o, 32);
  const float rstd = rsqrtf(q * (1.0f / 256.0f) + 1e-5f);

#pragma unroll
  for (int i = 0; i < 8; ++i) {
    int c = lane * 8 + i;
    float y = (v[i] - mean) * rstd * g[c] + bta[c];
    xr[c] = y;
    Xb[(long long)row * 256 + c] = (bf16_t)y;
  }
}

// ---------------------------------------------------------------------------
// Causal softmax of one row of S (f32, length T), scaled; bf16 probs to P.
// Block = 256 threads per row; columns > row get exactly 0.
// ---------------------------------------------------------------------------
__global__ void __launch_bounds__(256)
softmax_causal(const float* __restrict__ S, bf16_t* __restrict__ P,
               int T_, float scale)
{
  __shared__ float red[256];
  const int row = blockIdx.x;
  const int tid = threadIdx.x;
  const float* srow = S + (long long)row * T_;
  bf16_t*      prow = P + (long long)row * T_;

  float mx = -3.4e38f;
  for (int s = tid; s <= row; s += 256) mx = fmaxf(mx, srow[s] * scale);
  red[tid] = mx; __syncthreads();
  for (int o = 128; o > 0; o >>= 1) {
    if (tid < o) red[tid] = fmaxf(red[tid], red[tid + o]);
    __syncthreads();
  }
  mx = red[0]; __syncthreads();

  float sum = 0.f;
  for (int s = tid; s <= row; s += 256) sum += __expf(srow[s] * scale - mx);
  red[tid] = sum; __syncthreads();
  for (int o = 128; o > 0; o >>= 1) {
    if (tid < o) red[tid] += red[tid + o];
    __syncthreads();
  }
  const float inv = 1.0f / red[0];

  for (int s = tid; s < T_; s += 256) {
    float v = (s <= row) ? __expf(srow[s] * scale - mx) * inv : 0.0f;
    prow[s] = (bf16_t)v;
  }
}

// ---------------------------------------------------------------------------
extern "C" void kernel_launch(void* const* d_in, const int* in_sizes, int n_in,
                              void* d_out, int out_size, void* d_ws, size_t ws_size,
                              hipStream_t stream)
{
  (void)in_sizes; (void)n_in; (void)out_size; (void)ws_size;
  constexpr int Bz = 2, T = 2048, C = 256, H = 4, L = 4, FF = 1024, V = 32000;
  constexpr int BT = Bz * T;
  const float scale = 0.0625f;  // 1/sqrt(256)

  const int*   tokens = (const int*)  d_in[0];
  const float* embedW = (const float*)d_in[1];
  const float* posW   = (const float*)d_in[2];
  const float* ln1_g  = (const float*)d_in[3];
  const float* ln1_b  = (const float*)d_in[4];
  const float* wq     = (const float*)d_in[5];
  const float* wk     = (const float*)d_in[6];
  const float* wvw    = (const float*)d_in[7];
  const float* wo     = (const float*)d_in[8];
  const float* wo_b   = (const float*)d_in[9];
  const float* ln2_g  = (const float*)d_in[10];
  const float* ln2_b  = (const float*)d_in[11];
  const float* w1     = (const float*)d_in[12];
  const float* b1     = (const float*)d_in[13];
  const float* w2     = (const float*)d_in[14];
  const float* b2     = (const float*)d_in[15];
  const float* lnf_g  = (const float*)d_in[16];
  const float* lnf_b  = (const float*)d_in[17];
  const float* wf     = (const float*)d_in[18];
  const float* bfv    = (const float*)d_in[19];
  float* out = (float*)d_out;

  // --- workspace carve (all 256B aligned) ---
  char* p = (char*)d_ws;
  auto carve = [&](size_t bytes) -> char* {
    char* r = p; p += (bytes + 255) & ~(size_t)255; return r;
  };
  float*  x    = (float*) carve((size_t)BT * C * 4);       // residual stream (f32)
  bf16_t* xb   = (bf16_t*)carve((size_t)BT * C * 2);       // bf16 mirror of LN output
  bf16_t* q    = (bf16_t*)carve((size_t)H * BT * C * 2);   // [H][B*T][C]
  bf16_t* kx   = (bf16_t*)carve((size_t)H * BT * C * 2);   // [H][B*T][C]
  bf16_t* vtw  = (bf16_t*)carve((size_t)H * C * BT * 2);   // V^T: [H][C][B*T]
  bf16_t* ocat = (bf16_t*)carve((size_t)BT * H * C * 2);   // head-concat [B*T][H*C]
  float*  sc   = (float*) carve((size_t)T * T * 4);        // score slab (reused)
  bf16_t* pm   = (bf16_t*)carve((size_t)T * T * 2);        // softmax probs (bf16)
  bf16_t* hdn  = (bf16_t*)carve((size_t)BT * FF * 2);      // MLP hidden
  bf16_t* wq_t = (bf16_t*)carve((size_t)L * H * C * C * 2);
  bf16_t* wk_t = (bf16_t*)carve((size_t)L * H * C * C * 2);
  bf16_t* wv_t = (bf16_t*)carve((size_t)L * H * C * C * 2);
  bf16_t* wo_t = (bf16_t*)carve((size_t)L * C * (H * C) * 2);
  bf16_t* w1_t = (bf16_t*)carve((size_t)L * FF * C * 2);
  bf16_t* w2_t = (bf16_t*)carve((size_t)L * C * FF * 2);
  bf16_t* wf_t = (bf16_t*)carve((size_t)V * C * 2);

  auto cdiv = [](long long a, long long b) { return (int)((a + b - 1) / b); };

  // --- weight transpose + bf16 convert (B operands become N x K row-major) ---
  tconv<<<cdiv((long long)L*H*C*C, 256), 256, 0, stream>>>(wq,  wq_t, L*H, C,   C);
  tconv<<<cdiv((long long)L*H*C*C, 256), 256, 0, stream>>>(wk,  wk_t, L*H, C,   C);
  tconv<<<cdiv((long long)L*H*C*C, 256), 256, 0, stream>>>(wvw, wv_t, L*H, C,   C);
  tconv<<<cdiv((long long)L*(H*C)*C,256),256, 0, stream>>>(wo,  wo_t, L,   H*C, C);
  tconv<<<cdiv((long long)L*C*FF,  256), 256, 0, stream>>>(w1,  w1_t, L,   C,   FF);
  tconv<<<cdiv((long long)L*FF*C,  256), 256, 0, stream>>>(w2,  w2_t, L,   FF,  C);
  tconv<<<cdiv((long long)C*V,     256), 256, 0, stream>>>(wf,  wf_t, 1,   C,   V);

  // --- embedding + positional ---
  embed_add<<<cdiv((long long)BT * C, 256), 256, 0, stream>>>(
      tokens, embedW, posW, x, T, C, BT * C);

  auto gemm = [&](const bf16_t* A, int lda, const bf16_t* Bt, int ldb,
                  float* Cf, long long ldc, bf16_t* Cb, long long ldcb,
                  const float* bias, int M, int N, int K, int flags) {
    dim3 g(N / 128, M / 128);
    gemm_bf16_wmma<<<g, 256, 0, stream>>>(A, lda, Bt, ldb, Cf, ldc, Cb, ldcb,
                                          bias, K, flags);
  };

  for (int l = 0; l < L; ++l) {
    // x = LN1(x) in place; xb = bf16(x)
    layernorm_rows<<<BT / 8, 256, 0, stream>>>(x, xb, ln1_g + l * C, ln1_b + l * C);

    // Q, K, V projections per head (V stored transposed: [C][B*T])
    for (int h = 0; h < H; ++h) {
      const bf16_t* wqh = wq_t + (size_t)(l * H + h) * C * C;
      const bf16_t* wkh = wk_t + (size_t)(l * H + h) * C * C;
      const bf16_t* wvh = wv_t + (size_t)(l * H + h) * C * C;
      gemm(xb, C, wqh, C, nullptr, 0, q  + (size_t)h * BT * C, C, nullptr,
           BT, C, C, F_OUTB16);
      gemm(xb, C, wkh, C, nullptr, 0, kx + (size_t)h * BT * C, C, nullptr,
           BT, C, C, F_OUTB16);
      gemm(xb, C, wvh, C, nullptr, 0, vtw + (size_t)h * C * BT, BT, nullptr,
           BT, C, C, F_OUTB16T);
    }

    // attention per (batch, head), reusing the score slab
    for (int b = 0; b < Bz; ++b) {
      for (int h = 0; h < H; ++h) {
        const bf16_t* qbh = q  + ((size_t)h * BT + (size_t)b * T) * C;
        const bf16_t* kbh = kx + ((size_t)h * BT + (size_t)b * T) * C;
        // scores = Q K^T   (K row-major [T][C] is already the N x K operand)
        gemm(qbh, C, kbh, C, sc, T, nullptr, 0, nullptr, T, T, C, F_OUTF32);
        softmax_causal<<<T, 256, 0, stream>>>(sc, pm, T, scale);
        // O = P V, written straight into head-concat layout
        gemm(pm, T, vtw + (size_t)h * C * BT + (size_t)b * T, BT,
             nullptr, 0, ocat + (size_t)b * T * (H * C) + (size_t)h * C, H * C,
             nullptr, T, C, T, F_OUTB16);
      }
    }

    // x += ocat @ wo + wo_b
    gemm(ocat, H * C, wo_t + (size_t)l * C * (H * C), H * C, x, C,
         nullptr, 0, wo_b + l * C, BT, C, H * C, F_ADDRES | F_BIAS);

    // x = LN2(x); hdn = relu(xb @ w1 + b1); x += hdn @ w2 + b2
    layernorm_rows<<<BT / 8, 256, 0, stream>>>(x, xb, ln2_g + l * C, ln2_b + l * C);
    gemm(xb, C, w1_t + (size_t)l * FF * C, C, nullptr, 0, hdn, FF,
         b1 + l * FF, BT, FF, C, F_BIAS | F_RELU | F_OUTB16);
    gemm(hdn, FF, w2_t + (size_t)l * C * FF, FF, x, C,
         nullptr, 0, b2 + l * C, BT, C, FF, F_ADDRES | F_BIAS);
  }

  // logits = LNf(x) @ wf + bf
  layernorm_rows<<<BT / 8, 256, 0, stream>>>(x, xb, lnf_g, lnf_b);
  gemm(xb, C, wf_t, C, out, V, nullptr, 0, bfv, BT, V, C, F_OUTF32 | F_BIAS);
}
